// MultiheadCorrelation_43662637531401
// MI455X (gfx1250) — matrix operand order repaired and checked by reading, not compiled, not run on hardware
//
#include <hip/hip_runtime.h>

// Problem constants (reference: T=4096, B=4, D=1024, H=16 -> hd=64)
#define T_SEQ   4096
#define B_SZ    4
#define D_MODEL 1024
#define NHEAD   16
#define HD      64
#define MROWS   (T_SEQ * B_SZ)     // 16384
#define QKV_N   (3 * D_MODEL)      // 3072
#define NBH     (B_SZ * NHEAD)     // 64

typedef __bf16 bf16_t;
typedef __attribute__((ext_vector_type(16))) __bf16 v16bf;
typedef __attribute__((ext_vector_type(8)))  __bf16 v8bf;
typedef __attribute__((ext_vector_type(8)))  float  v8f;
typedef __attribute__((ext_vector_type(4)))  unsigned int u32x4;
typedef __attribute__((ext_vector_type(8)))  unsigned int u32x8;

__device__ __forceinline__ unsigned short f2bf(float f) {
  union { float f; unsigned u; } v; v.f = f;
  unsigned r = v.u + 0x7fffu + ((v.u >> 16) & 1u);   // round to nearest even
  return (unsigned short)(r >> 16);
}
__device__ __forceinline__ float bf2f(unsigned short h) {
  union { unsigned u; float f; } v; v.u = ((unsigned)h) << 16; return v.f;
}

// ---- WMMA fragment loaders (16-bit A/B, contiguous k = 16*(lane/16) + s, same order for A and B) ----
// A (16x32, MxK): lanes 0-15 row M=lane, K = 0..15; lanes 16-31 K = 16..31
__device__ __forceinline__ v16bf frag_a(const bf16_t* p, int lane) {
  const int base = (lane < 16) ? 0 : 16;
  v8bf lo = *(const v8bf*)(p + base);
  v8bf hi = *(const v8bf*)(p + base + 8);
  v16bf a;
#pragma unroll
  for (int i = 0; i < 8; ++i) { a[i] = lo[i]; a[i + 8] = hi[i]; }
  return a;
}
// B (32x16, KxN): lane gives N=lane&15; lanes 0-15 hold K=k0..k0+15, lanes 16-31 K=k0+16..k0+31
__device__ __forceinline__ v16bf frag_b(const bf16_t* p, int lane) {
  const int base = (lane < 16) ? 0 : 16;
  v8bf lo = *(const v8bf*)(p + base);
  v8bf hi = *(const v8bf*)(p + base + 8);
  v16bf b;
#pragma unroll
  for (int i = 0; i < 8; ++i) { b[i] = lo[i]; b[i + 8] = hi[i]; }
  return b;
}
__device__ __forceinline__ v8f wmma_bf16(v16bf a, v16bf b, v8f c) {
  return __builtin_amdgcn_wmma_f32_16x16x32_bf16(false, a, false, b, (short)0, c,
                                                 false, false);
}

// ---- Tensor Data Mover: 2D tile (32 rows x 64 bf16) global -> LDS ----------
// D# per CDNA5 ISA: group0 = {count, lds_addr, global_addr,
// type=2}; group1 = {data_size=2B, tensor dims = tile dims (32x64), dim0
// stride}. Issued wave-uniform via inline asm (portable across toolchains),
// tracked by TENSORcnt.
__device__ __forceinline__ void tdm_load_tile_32x64(unsigned lds_off,
                                                    const void* gptr,
                                                    unsigned stride0_elems) {
  const unsigned long long ga = (unsigned long long)(uintptr_t)gptr;
  u32x4 g0;
  g0[0] = 1u;                                             // count=1, user mode
  g0[1] = lds_off;                                        // lds_addr (bytes)
  g0[2] = (unsigned)(ga & 0xFFFFFFFFu);                   // global_addr[31:0]
  g0[3] = (unsigned)((ga >> 32) & 0x01FFFFFFu) | 0x80000000u; // [56:32] | type=2
  u32x8 g1;
  g1[0] = 0x00010000u;              // workgroup_mask=0, data_size=1 (2 bytes)
  g1[1] = 64u << 16;                // tensor_dim0 = 64 (low half)
  g1[2] = 32u << 16;                // tensor_dim0 hi=0 | tensor_dim1 = 32 (lo)
  g1[3] = 64u << 16;                // tensor_dim1 hi=0 | tile_dim0 = 64
  g1[4] = 32u;                      // tile_dim1 = 32, tile_dim2 = 0
  g1[5] = stride0_elems;            // tensor_dim0_stride[31:0]
  g1[6] = 0u;                       // stride0[47:32] | stride1[15:0]
  g1[7] = 0u;                       // stride1[47:16]
  asm volatile("tensor_load_to_lds %0, %1" :: "s"(g0), "s"(g1) : "memory");
}

// ---------------- fp32 -> bf16 conversion ----------------
__global__ void cvt_f32_bf16_kernel(const float* __restrict__ src,
                                    unsigned short* __restrict__ dst, int n) {
  for (int i = blockIdx.x * blockDim.x + threadIdx.x; i < n;
       i += gridDim.x * blockDim.x)
    dst[i] = f2bf(src[i]);
}

__global__ void zero_kernel(float* __restrict__ p, int n) {
  const int i = blockIdx.x * blockDim.x + threadIdx.x;
  if (i < n) p[i] = 0.0f;
}

// ---------------- generic WMMA GEMM: out(M,N) = A(M,K) @ W(N,K)^T + bias ----
// One wave computes a 32x64 tile: 2 A fragments x 4 B fragments -> 8
// independent WMMA chains per K-step. K is a compile-time constant so the
// second A row and the 3 extra W rows live at constant byte offsets
// (16*K*2 = 32KB, within the 24-bit instruction offset) from two base
// pointers -> fewer live address VGPRs, no spills in the hot loop.
template <int K, bool STORE_BF16>
__global__ void gemm_bf16_wmma_kernel(const unsigned short* __restrict__ A,
                                      const unsigned short* __restrict__ W,
                                      const float* __restrict__ bias,
                                      void* __restrict__ out,
                                      int M, int N) {
  const int lane  = threadIdx.x & 31;
  const int gwave = blockIdx.x * (blockDim.x >> 5) + (threadIdx.x >> 5);
  const int tilesM = M >> 5;
  const int tilesN = (N + 63) >> 6;
  if (gwave >= tilesM * tilesN) return;           // wave-uniform: EXEC stays full
  const int tm = gwave % tilesM;
  const int tn = gwave / tilesM;

  const int am0 = (tm << 5) + (lane & 15);        // A rows, first 16-row half
  const int wn  = (tn << 6) + (lane & 15);
  const bf16_t* Ap = (const bf16_t*)A + (size_t)am0 * K;  // row base (A)
  const bf16_t* Wp = (const bf16_t*)W + (size_t)wn * K;   // row base (W)
  const int R16 = 16 * K;                         // 16-row stride (elements)

  v8f acc[2][4] = {};
  for (int k0 = 0; k0 < K; k0 += 32) {
    // branchless distant prefetch (overshoot stays inside workspace; bad
    // translations are silently dropped for speculative prefetch)
    __builtin_prefetch(Ap + k0 + 256, 0, 1);
    v16bf a0 = frag_a(Ap + k0,            lane);
    v16bf a1 = frag_a(Ap + 1 * R16 + k0,  lane);
    v16bf b0 = frag_b(Wp + k0,            lane);
    v16bf b1 = frag_b(Wp + 1 * R16 + k0,  lane);
    v16bf b2 = frag_b(Wp + 2 * R16 + k0,  lane);
    v16bf b3 = frag_b(Wp + 3 * R16 + k0,  lane);
    acc[0][0] = wmma_bf16(a0, b0, acc[0][0]);
    acc[0][1] = wmma_bf16(a0, b1, acc[0][1]);
    acc[0][2] = wmma_bf16(a0, b2, acc[0][2]);
    acc[0][3] = wmma_bf16(a0, b3, acc[0][3]);
    acc[1][0] = wmma_bf16(a1, b0, acc[1][0]);
    acc[1][1] = wmma_bf16(a1, b1, acc[1][1]);
    acc[1][2] = wmma_bf16(a1, b2, acc[1][2]);
    acc[1][3] = wmma_bf16(a1, b3, acc[1][3]);
  }

  // C/D layout: lane L, VGPR r -> row = r + (L>=16 ? 8:0), col = L&15
#pragma unroll
  for (int half = 0; half < 2; ++half) {
    const int mrow = (tm << 5) + (half << 4) + ((lane >= 16) ? 8 : 0);
#pragma unroll
    for (int sub = 0; sub < 4; ++sub) {
      const int ncol = (tn << 6) + (sub << 4) + (lane & 15);
      const float bv = bias[ncol];
#pragma unroll
      for (int r = 0; r < 8; ++r) {
        const float val = acc[half][sub][r] + bv;
        const size_t idx = (size_t)(mrow + r) * N + ncol;
        if (STORE_BF16) ((unsigned short*)out)[idx] = f2bf(val);
        else            ((float*)out)[idx] = val;
      }
    }
  }
}

// ---------------- per-(b,h,d) sums over T (parallel over T, LDS + atomics) --
// stats layout: [0)=sumq, [4096)=ssq, [8192)=sumk, [12288)=ssk
__global__ void stats_kernel(const unsigned short* __restrict__ qkv,
                             float* __restrict__ stats) {
  const int blk = blockIdx.x;                // NBH * 16 blocks
  const int bh = blk >> 4, tc = blk & 15;    // 16 chunks of 256 t
  const int b = bh >> 4, h = bh & 15;
  const int tid = threadIdx.x;               // 256
  const int d = tid & 63, ts = tid >> 6;     // ts = 0..3
  const int colq = h * HD + d, colk = D_MODEL + colq;
  float sq = 0.f, s2q = 0.f, sk = 0.f, s2k = 0.f;
  for (int i = 0; i < 64; ++i) {
    const int t = (tc << 8) + (i << 2) + ts;
    const size_t base = (size_t)(t * B_SZ + b) * QKV_N;
    const float qv = bf2f(qkv[base + colq]);
    const float kv = bf2f(qkv[base + colk]);
    sq += qv; s2q += qv * qv; sk += kv; s2k += kv * kv;
  }
  __shared__ float r0[256], r1[256], r2[256], r3[256];
  r0[tid] = sq; r1[tid] = s2q; r2[tid] = sk; r3[tid] = s2k;
  __syncthreads();
  if (ts == 0) {
    float a0 = 0.f, a1 = 0.f, a2 = 0.f, a3 = 0.f;
#pragma unroll
    for (int j = 0; j < 4; ++j) {
      a0 += r0[j * 64 + d]; a1 += r1[j * 64 + d];
      a2 += r2[j * 64 + d]; a3 += r3[j * 64 + d];
    }
    atomicAdd(&stats[            bh * HD + d], a0);
    atomicAdd(&stats[ 1 * 4096 + bh * HD + d], a1);
    atomicAdd(&stats[ 2 * 4096 + bh * HD + d], a2);
    atomicAdd(&stats[ 3 * 4096 + bh * HD + d], a3);
  }
}

// ---------------- Gram: G[bh][d][e] = sum_t q[t,d] * k[t,e] -----------------
// TDM stages each 32x64 bf16 tile of q and k straight into LDS (no VGPR
// round-trip); waves gather WMMA fragments from the t-major tile with LDS
// reads using the exact A/B lane->K mappings. One 128-thread block per (b,h).
__global__ void gram_kernel(const unsigned short* __restrict__ qkv,
                            float* __restrict__ gram) {
  __shared__ __attribute__((aligned(16))) unsigned short qS[32 * HD];
  __shared__ __attribute__((aligned(16))) unsigned short kS[32 * HD];
  const int bh = blockIdx.x, b = bh >> 4, h = bh & 15;
  const int tid = threadIdx.x, wave = tid >> 5, lane = tid & 31;
  const bf16_t* qSb = (const bf16_t*)qS;
  const bf16_t* kSb = (const bf16_t*)kS;

  v8f acc[4] = {};
  const int am = (wave << 4) + (lane & 15);   // head-dim row of q (A row)
  const int n0 = lane & 15;

  for (int t0 = 0; t0 < T_SEQ; t0 += 32) {
    if (wave == 0) {
      const size_t base = (size_t)(t0 * B_SZ + b) * QKV_N + h * HD;
      tdm_load_tile_32x64((unsigned)(uintptr_t)qS, qkv + base, B_SZ * QKV_N);
      tdm_load_tile_32x64((unsigned)(uintptr_t)kS, qkv + base + D_MODEL,
                          B_SZ * QKV_N);
      __builtin_amdgcn_s_wait_tensorcnt(0);
    }
    __syncthreads();

    v16bf a, bf0, bf1, bf2, bf3;
#pragma unroll
    for (int i = 0; i < 16; ++i) {
      const int ka = i + ((lane < 16) ? 0 : 16);                     // A K-map
      a[i] = qSb[ka * HD + am];
      const int kb = i + ((lane < 16) ? 0 : 16);                     // B K-map
      bf0[i] = kSb[kb * HD + n0 +  0];
      bf1[i] = kSb[kb * HD + n0 + 16];
      bf2[i] = kSb[kb * HD + n0 + 32];
      bf3[i] = kSb[kb * HD + n0 + 48];
    }
    acc[0] = wmma_bf16(a, bf0, acc[0]);
    acc[1] = wmma_bf16(a, bf1, acc[1]);
    acc[2] = wmma_bf16(a, bf2, acc[2]);
    acc[3] = wmma_bf16(a, bf3, acc[3]);
    __syncthreads();
  }

  const int mrow = (wave << 4) + ((lane >= 16) ? 8 : 0);
#pragma unroll
  for (int sub = 0; sub < 4; ++sub) {
    const int ncol = (sub << 4) + (lane & 15);
#pragma unroll
    for (int r = 0; r < 8; ++r)
      gram[(size_t)bh * (HD * HD) + (size_t)(mrow + r) * HD + ncol] = acc[sub][r];
  }
}

// ---------------- corr = clip(cov/std) ; stored TRANSPOSED (e-major) --------
__global__ void corr_kernel(const float* __restrict__ gram,
                            const float* __restrict__ stats,
                            unsigned short* __restrict__ corrT) {
  const int idx = blockIdx.x * blockDim.x + threadIdx.x;   // 64*64*64
  if (idx >= NBH * HD * HD) return;
  const int bh = idx >> 12, rem = idx & 4095, d = rem >> 6, e = rem & 63;
  const float invT = 1.0f / (float)T_SEQ;
  const float mq = stats[           bh * HD + d] * invT;
  const float mk = stats[2 * 4096 + bh * HD + e] * invT;
  const float cov = gram[idx] - (float)T_SEQ * mq * mk;
  const float sx = stats[1 * 4096 + bh * HD + d] - (float)T_SEQ * mq * mq;
  const float sy = stats[3 * 4096 + bh * HD + e] - (float)T_SEQ * mk * mk;
  float c = cov * rsqrtf(fmaxf(sx * sy, 1e-30f));
  c = fminf(1.0f, fmaxf(-1.0f, c));
  corrT[(size_t)bh * (HD * HD) + (size_t)e * HD + d] = f2bf(c);   // [e][d]
}

// ---------------- attn[t,b,h,:] = v[t,b,h,:] @ corr[bh] --------------------
__global__ void attn_kernel(const unsigned short* __restrict__ qkv,
                            const unsigned short* __restrict__ corrT,
                            unsigned short* __restrict__ attn) {
  const int lane  = threadIdx.x & 31;
  const int gwave = blockIdx.x * (blockDim.x >> 5) + (threadIdx.x >> 5);
  const int bh = gwave >> 8;                 // 256 t-tiles per (b,h)
  const int tile = gwave & 255;
  if (bh >= NBH) return;
  const int b = bh >> 4, h = bh & 15;

  const bf16_t* Vb = (const bf16_t*)qkv;
  const bf16_t* Cb = (const bf16_t*)corrT;
  const int at = (tile << 4) + (lane & 15);
  const size_t vrow = (size_t)(at * B_SZ + b) * QKV_N + 2 * D_MODEL + h * HD;

  v8f acc[4] = {};
#pragma unroll
  for (int k0 = 0; k0 < HD; k0 += 32) {
    v16bf a = frag_a(Vb + vrow + k0, lane);
#pragma unroll
    for (int sub = 0; sub < 4; ++sub) {
      const int n = (sub << 4) + (lane & 15);
      v16bf bfr = frag_b(Cb + (size_t)bh * (HD * HD) + (size_t)n * HD + k0, lane);
      acc[sub] = wmma_bf16(a, bfr, acc[sub]);
    }
  }

  const int trow = (tile << 4) + ((lane >= 16) ? 8 : 0);
#pragma unroll
  for (int sub = 0; sub < 4; ++sub) {
    const int ncol = (sub << 4) + (lane & 15);
#pragma unroll
    for (int r = 0; r < 8; ++r)
      attn[(size_t)((trow + r) * B_SZ + b) * D_MODEL + h * HD + ncol] =
          f2bf(acc[sub][r]);
  }
}

// ============================== host side ===================================
extern "C" void kernel_launch(void* const* d_in, const int* in_sizes, int n_in,
                              void* d_out, int out_size, void* d_ws, size_t ws_size,
                              hipStream_t stream) {
  const float* x    = (const float*)d_in[0];   // (T,B,D)
  const float* Wqkv = (const float*)d_in[1];   // (3D,D)
  const float* bqkv = (const float*)d_in[2];   // (3D)
  const float* Wout = (const float*)d_in[3];   // (D,D)
  const float* bout = (const float*)d_in[4];   // (D)
  float* out = (float*)d_out;                  // (T,B,D) fp32

  char* ws = (char*)d_ws;
  size_t off = 0;
  auto alloc = [&](size_t bytes) { void* p = ws + off; off = (off + bytes + 255) & ~(size_t)255; return p; };
  unsigned short* x_bf    = (unsigned short*)alloc((size_t)MROWS * D_MODEL * 2);
  unsigned short* wqkv_bf = (unsigned short*)alloc((size_t)QKV_N * D_MODEL * 2);
  unsigned short* wout_bf = (unsigned short*)alloc((size_t)D_MODEL * D_MODEL * 2);
  unsigned short* qkv_bf  = (unsigned short*)alloc((size_t)MROWS * QKV_N * 2);
  unsigned short* attn_bf = (unsigned short*)alloc((size_t)MROWS * D_MODEL * 2);
  float* gram  = (float*)alloc((size_t)NBH * HD * HD * 4);
  float* stats = (float*)alloc((size_t)4 * NBH * HD * 4);
  unsigned short* corrT = (unsigned short*)alloc((size_t)NBH * HD * HD * 2);
  (void)ws_size; (void)n_in; (void)in_sizes; (void)out_size;

  // 1) fp32 -> bf16 + zero the stats accumulators
  cvt_f32_bf16_kernel<<<1024, 256, 0, stream>>>(x, x_bf, MROWS * D_MODEL);
  cvt_f32_bf16_kernel<<<256, 256, 0, stream>>>(Wqkv, wqkv_bf, QKV_N * D_MODEL);
  cvt_f32_bf16_kernel<<<256, 256, 0, stream>>>(Wout, wout_bf, D_MODEL * D_MODEL);
  zero_kernel<<<64, 256, 0, stream>>>(stats, 4 * NBH * HD);

  // 2) qkv = x @ Wqkv^T + bqkv  (M=16384, N=3072, K=1024), bf16 out
  {
    const int waves = (MROWS / 32) * (QKV_N / 64);
    gemm_bf16_wmma_kernel<D_MODEL, true><<<(waves + 7) / 8, 256, 0, stream>>>(
        x_bf, wqkv_bf, bqkv, qkv_bf, MROWS, QKV_N);
  }

  // 3) per-(b,h,d) sums / sums of squares over T
  stats_kernel<<<NBH * 16, 256, 0, stream>>>(qkv_bf, stats);

  // 4) Gram matrices: TDM-staged LDS tiles + WMMA
  gram_kernel<<<NBH, 128, 0, stream>>>(qkv_bf, gram);

  // 5) Pearson correlation (clipped), stored transposed as bf16
  corr_kernel<<<(NBH * HD * HD + 255) / 256, 256, 0, stream>>>(gram, stats, corrT);

  // 6) attn = v @ corr
  {
    const int waves = NBH * (T_SEQ / 16);
    attn_kernel<<<(waves + 7) / 8, 256, 0, stream>>>(qkv_bf, corrT, attn_bf);
  }

  // 7) out = attn @ Wout^T + bout (fp32 store to d_out)
  {
    const int waves = (MROWS / 32) * (D_MODEL / 64);
    gemm_bf16_wmma_kernel<D_MODEL, false><<<(waves + 7) / 8, 256, 0, stream>>>(
        attn_bf, wout_bf, bout, out, MROWS, D_MODEL);
  }
}
